// ContrastiveLoss_13683765805546
// MI455X (gfx1250) — matrix operand order repaired and checked by reading, not compiled, not run on hardware
//
#include <hip/hip_runtime.h>
#include <math.h>
#include <stdint.h>

// Problem constants (from the reference)
#define N_PAIR   4096
#define TWO_N    8192
#define DIM      256
#define INV_T    10.0f           // 1 / TEMPERATURE

// Tiling
#define KB              8                       // 256 / 32 K-steps
#define COL_SPLITS      8
#define COLS_PER_SPLIT  (TWO_N / COL_SPLITS)    // 1024
#define TILE_COLS       64                      // columns staged in LDS per step
#define TILES_PER_SPLIT (COLS_PER_SPLIT / TILE_COLS)  // 16
#define SUBTILES        (TILE_COLS / 16)        // 4
#define WAVES_PER_BLK   8
#define ROWS_PER_WAVE   32                      // two 16-row WMMA tiles
#define ROWS_PER_BLK    (WAVES_PER_BLK * ROWS_PER_WAVE)  // 256
#define ROW_BLOCKS      (TWO_N / ROWS_PER_BLK)  // 32
#define LDSROW          264                     // 256 f16 + 8 f16 pad (528 B row)

typedef _Float16 v16h __attribute__((ext_vector_type(16)));
typedef _Float16 h8   __attribute__((ext_vector_type(8)));
typedef float    v8f  __attribute__((ext_vector_type(8)));
typedef float    f4   __attribute__((ext_vector_type(4)));

// ---------------------------------------------------------------------------
// Kernel 1: L2-normalize each of the 2N rows (z_i stacked over z_j), emit f16.
// ---------------------------------------------------------------------------
__global__ void __launch_bounds__(256) nrm_kernel(const float* __restrict__ zi,
                                                  const float* __restrict__ zj,
                                                  _Float16* __restrict__ repsH) {
    int wave = (blockIdx.x * blockDim.x + threadIdx.x) >> 5;
    int lane = threadIdx.x & 31;
    if (wave >= TWO_N) return;

    const float* src = (wave < N_PAIR) ? (zi + (size_t)wave * DIM)
                                       : (zj + (size_t)(wave - N_PAIR) * DIM);
    f4 a = *(const f4*)(src + lane * 8);
    f4 b = *(const f4*)(src + lane * 8 + 4);

    float ss = a.x*a.x + a.y*a.y + a.z*a.z + a.w*a.w
             + b.x*b.x + b.y*b.y + b.z*b.z + b.w*b.w;
    #pragma unroll
    for (int off = 16; off > 0; off >>= 1) ss += __shfl_xor(ss, off, 32);

    float inv = rsqrtf(ss);

    h8 o;
    o[0] = (_Float16)(a.x * inv); o[1] = (_Float16)(a.y * inv);
    o[2] = (_Float16)(a.z * inv); o[3] = (_Float16)(a.w * inv);
    o[4] = (_Float16)(b.x * inv); o[5] = (_Float16)(b.y * inv);
    o[6] = (_Float16)(b.z * inv); o[7] = (_Float16)(b.w * inv);
    *(h8*)(repsH + (size_t)wave * DIM + lane * 8) = o;
}

// ---------------------------------------------------------------------------
// Async DMA: global -> LDS, 16 B per lane, tracked by ASYNCcnt.
// ---------------------------------------------------------------------------
__device__ __forceinline__ void async_g2lds_b128(unsigned lds_off, uint64_t gaddr) {
    asm volatile("global_load_async_to_lds_b128 %0, %1, off"
                 :: "v"(lds_off), "v"(gaddr)
                 : "memory");
}

// Stage TILE_COLS rows of R (512 B each) into the padded LDS buffer.
// 8 waves x 8 ops x (32 lanes x 16 B) = 32 KiB.
__device__ __forceinline__ void stage_tile(const _Float16* __restrict__ R,
                                           int colTileBase, unsigned lbase,
                                           int w, int lane) {
    uint64_t gbase = (uint64_t)(const void*)(R + (size_t)colTileBase * DIM);
    #pragma unroll
    for (int j = 0; j < 8; ++j) {
        const int seg = w * 8 + j;                  // staged row 0..63
        async_g2lds_b128(lbase + (unsigned)(seg * (LDSROW * 2) + lane * 16),
                         gbase + (uint64_t)seg * (DIM * 2) + (uint64_t)lane * 16);
    }
}

// Load one 16-row A-fragment set (8 x v16h) for rows [rowBase, rowBase+16).
// ISA 16-bit A 16x32: lane<16 -> K {0..7,16..23}; lane>=16 -> +8.
__device__ __forceinline__ void load_a_frags(const _Float16* __restrict__ R,
                                             int rowBase, int mrow, int hi,
                                             v16h* aF) {
    const _Float16* arow = R + (size_t)(rowBase + mrow) * DIM;
    #pragma unroll
    for (int kb = 0; kb < KB; ++kb) {
        const int k0 = kb * 32 + hi * 8;
        h8 lo = *(const h8*)(arow + k0);
        h8 hh = *(const h8*)(arow + k0 + 16);
        v16h a;
        #pragma unroll
        for (int i = 0; i < 8; ++i) { a[i] = lo[i]; a[8 + i] = hh[i]; }
        aF[kb] = a;
    }
}

// ---------------------------------------------------------------------------
// Kernel 2: fused sim-GEMM + masked exp-sum, WMMA f16->f32, async-LDS staged B.
// Block: 8 waves x 32 rows = 256 rows; sweeps one 1024-column split in 64-col
// LDS tiles (double buffered). Each B fragment read from LDS feeds TWO
// independent WMMAs (two 16-row tiles per wave) -> 512 B LDS per WMMA.
// Per-row Sigma exp(10*dot - 10), diagonal masked, written to
// partial[colSplit][row] (no atomics -> deterministic).
// ---------------------------------------------------------------------------
__global__ void __launch_bounds__(256) simlse_kernel(const _Float16* __restrict__ R,
                                                     float* __restrict__ partial) {
    __shared__ _Float16 smem[2][TILE_COLS * LDSROW];   // 2 x 33 KiB

    const int lane = threadIdx.x & 31;
    const int w    = threadIdx.x >> 5;
    const int rowBase0 = blockIdx.x * ROWS_PER_BLK + w * ROWS_PER_WAVE;
    const int rowBase1 = rowBase0 + 16;
    const int cs       = blockIdx.y;
    const int colBase0 = cs * COLS_PER_SPLIT;

    const int mrow = lane & 15;       // A row / B column within a 16-tile
    const int hi   = lane >> 4;       // lane half selector

    // Two resident A-fragment sets (rows rowBase0.. and rowBase1..).
    v16h aF0[KB], aF1[KB];
    load_a_frags(R, rowBase0, mrow, hi, aF0);
    load_a_frags(R, rowBase1, mrow, hi, aF1);

    float acc0[8], acc1[8];
    #pragma unroll
    for (int v = 0; v < 8; ++v) { acc0[v] = 0.0f; acc1[v] = 0.0f; }

    const unsigned lbase0 = (unsigned)(uintptr_t)&smem[0][0];
    const unsigned lbase1 = (unsigned)(uintptr_t)&smem[1][0];

    // Prime the pipeline: stage tile 0 into buffer 0.
    stage_tile(R, colBase0, lbase0, w, lane);

    for (int t = 0; t < TILES_PER_SPLIT; ++t) {
        const int cur = t & 1;

        // Wait for this wave's async stores, then sync so all 8 waves' staged
        // segments are visible.
        asm volatile("s_wait_asynccnt 0x0" ::: "memory");
        __syncthreads();

        // Kick off the DMA for the next tile while we compute this one.
        if (t + 1 < TILES_PER_SPLIT)
            stage_tile(R, colBase0 + (t + 1) * TILE_COLS,
                       cur ? lbase0 : lbase1, w, lane);

        const _Float16* sbase = &smem[cur][0];
        #pragma unroll
        for (int sub = 0; sub < SUBTILES; ++sub) {
            // B layout (16-bit B 32x16): lane holds column (lane&15),
            // K {0..7,16..23} + hi*8 (matches A).
            const _Float16* bptr = sbase + (size_t)(sub * 16 + mrow) * LDSROW;

            v16h bF[KB];
            #pragma unroll
            for (int kb = 0; kb < KB; ++kb) {
                const _Float16* p = bptr + kb * 32 + hi * 8;
                h8 lo = *(const h8*)p;
                h8 hh = *(const h8*)(p + 16);
                v16h b;
                #pragma unroll
                for (int i = 0; i < 8; ++i) { b[i] = lo[i]; b[8 + i] = hh[i]; }
                bF[kb] = b;
            }

            v8f c0 = {};
            v8f c1 = {};
            #pragma unroll
            for (int kb = 0; kb < KB; ++kb) {
                c0 = __builtin_amdgcn_wmma_f32_16x16x32_f16(
                         false, aF0[kb], false, bF[kb], (short)0, c0, false, false);
                c1 = __builtin_amdgcn_wmma_f32_16x16x32_f16(
                         false, aF1[kb], false, bF[kb], (short)0, c1, false, false);
            }

            // e = exp(sim - 10), sim = 10*dot, |dot|<=1 -> exponent <= 0.
            // C layout: c[v] = row rowBase+v+8*hi, col = tile col + (lane&15).
            const int col = colBase0 + t * TILE_COLS + sub * 16 + mrow;
            #pragma unroll
            for (int v = 0; v < 8; ++v) {
                const int r0 = rowBase0 + v + hi * 8;
                const int r1 = rowBase1 + v + hi * 8;
                const float e0 = __expf(c0[v] * INV_T - INV_T);
                const float e1 = __expf(c1[v] * INV_T - INV_T);
                acc0[v] += (r0 != col) ? e0 : 0.0f;
                acc1[v] += (r1 != col) ? e1 : 0.0f;
            }
        }
        // All waves done reading `cur` before iteration t+1 stages into it.
        __syncthreads();
    }

    // Reduce across the 16 lanes sharing each row (xor 1,2,4,8 stays in-half).
    #pragma unroll
    for (int v = 0; v < 8; ++v) {
        float s0 = acc0[v];
        float s1 = acc1[v];
        #pragma unroll
        for (int off = 1; off < 16; off <<= 1) {
            s0 += __shfl_xor(s0, off, 32);
            s1 += __shfl_xor(s1, off, 32);
        }
        acc0[v] = s0;
        acc1[v] = s1;
    }

    if ((lane & 15) == 0) {
        float* dst0 = partial + (size_t)cs * TWO_N + rowBase0 + hi * 8;
        float* dst1 = partial + (size_t)cs * TWO_N + rowBase1 + hi * 8;
        #pragma unroll
        for (int v = 0; v < 8; ++v) { dst0[v] = acc0[v]; dst1[v] = acc1[v]; }
    }
}

// ---------------------------------------------------------------------------
// Kernel 3: positive-pair logits. pos[i] = pos[i+N] = 10 * dot(r_i, r_{i+N}).
// ---------------------------------------------------------------------------
__global__ void __launch_bounds__(256) pos_kernel(const _Float16* __restrict__ R,
                                                  float* __restrict__ pos) {
    int wave = (blockIdx.x * blockDim.x + threadIdx.x) >> 5;
    int lane = threadIdx.x & 31;
    if (wave >= N_PAIR) return;

    h8 a = *(const h8*)(R + (size_t)wave * DIM + lane * 8);
    h8 b = *(const h8*)(R + (size_t)(wave + N_PAIR) * DIM + lane * 8);
    float s = 0.0f;
    #pragma unroll
    for (int i = 0; i < 8; ++i) s += (float)a[i] * (float)b[i];
    #pragma unroll
    for (int off = 16; off > 0; off >>= 1) s += __shfl_xor(s, off, 32);

    if (lane == 0) {
        float p = s * INV_T;
        pos[wave] = p;
        pos[wave + N_PAIR] = p;
    }
}

// ---------------------------------------------------------------------------
// Kernel 4: loss = mean_r( 10 + log(Sigma_cs partial[cs][r]) - pos[r] ).
// ---------------------------------------------------------------------------
__global__ void __launch_bounds__(256) fin_kernel(const float* __restrict__ partial,
                                                  const float* __restrict__ pos,
                                                  float* __restrict__ out) {
    __shared__ float red[256];
    const int t = threadIdx.x;
    float s = 0.0f;
    for (int r = t; r < TWO_N; r += 256) {
        float S = 0.0f;
        #pragma unroll
        for (int c = 0; c < COL_SPLITS; ++c) S += partial[(size_t)c * TWO_N + r];
        s += INV_T + logf(S) - pos[r];
    }
    red[t] = s;
    __syncthreads();
    #pragma unroll
    for (int off = 128; off > 0; off >>= 1) {
        if (t < off) red[t] += red[t + off];
        __syncthreads();
    }
    if (t == 0) out[0] = red[0] / (float)TWO_N;
}

// ---------------------------------------------------------------------------
extern "C" void kernel_launch(void* const* d_in, const int* in_sizes, int n_in,
                              void* d_out, int out_size, void* d_ws, size_t ws_size,
                              hipStream_t stream) {
    const float* zi = (const float*)d_in[0];
    const float* zj = (const float*)d_in[1];
    float* out = (float*)d_out;

    char* ws = (char*)d_ws;
    _Float16* repsH = (_Float16*)ws;                                   // 4 MiB
    float* partial  = (float*)(ws + (size_t)TWO_N * DIM * 2);          // 256 KiB
    float* pos      = (float*)(ws + (size_t)TWO_N * DIM * 2
                                  + (size_t)COL_SPLITS * TWO_N * 4);   // 32 KiB

    // 1) normalize -> f16 reps
    nrm_kernel<<<TWO_N / WAVES_PER_BLK, 256, 0, stream>>>(zi, zj, repsH);
    // 2) positive logits
    pos_kernel<<<N_PAIR / WAVES_PER_BLK, 256, 0, stream>>>(repsH, pos);
    // 3) fused WMMA similarity + masked exp-sum partials (async-LDS staged)
    dim3 grid(ROW_BLOCKS, COL_SPLITS);
    simlse_kernel<<<grid, 256, 0, stream>>>(repsH, partial);
    // 4) final logsumexp assembly + mean
    fin_kernel<<<1, 256, 0, stream>>>(partial, pos, out);
}
